// AxialSelfAttention2d_11295763988982
// MI455X (gfx1250) — compile-verified, matrix operand order checked by reading, not run on hardware
//
#include <hip/hip_runtime.h>
#include <stdint.h>

typedef __attribute__((ext_vector_type(16))) __bf16 v16bf;
typedef __attribute__((ext_vector_type(8)))  float  v8f;

#define HEADS 12
#define CDIM  32
#define EMB   384
#define SDIM  128
#define LDIM  256
#define NPOS  (SDIM * LDIM)   // 32768
#define K3    (3 * EMB)       // 1152
#define LN_EPS 1e-5f

// ---------------------------------------------------------------------------
// Fragment helpers.
// 16-bit A/B fragment striping (CDNA5 ISA 7.12.2): lanes 0-15 hold K in
// {0..7, 16..23}; lanes 16-31 hold K in {8..15, 24..31}. With rows stored
// contiguously, each lane needs two 16-byte chunks at +klo and +klo+16.
// Works for both global and LDS (generic) pointers.
// ---------------------------------------------------------------------------
__device__ __forceinline__ v16bf load_frag(const __bf16* row, int klo) {
    union { uint4 u[2]; v16bf v; } t;
    t.u[0] = *reinterpret_cast<const uint4*>(row + klo);
    t.u[1] = *reinterpret_cast<const uint4*>(row + klo + 16);
    return t.v;
}

__device__ __forceinline__ v8f vzero8() {
    v8f z = {0.f, 0.f, 0.f, 0.f, 0.f, 0.f, 0.f, 0.f};
    return z;
}

__device__ __forceinline__ v8f wmma_bf16(v16bf a, v16bf b, v8f c) {
    return __builtin_amdgcn_wmma_f32_16x16x32_bf16(
        /*neg_a=*/false, a, /*neg_b=*/false, b,
        /*c_mod=*/(short)0, c, /*reuse_a=*/false, /*reuse_b=*/false);
}

// ---------------------------------------------------------------------------
// Async global->LDS copy (CDNA5, ASYNCcnt-tracked; no VGPR data movement).
// lds_byte_off: workgroup-relative LDS byte offset (low 32 bits of a generic
// LDS pointer per ISA 10.2 aperture mapping). 16 bytes per lane.
// ---------------------------------------------------------------------------
__device__ __forceinline__ void async_copy_b128(uint32_t lds_byte_off, const void* gptr) {
    asm volatile("global_load_async_to_lds_b128 %0, %1, off"
                 :: "v"(lds_byte_off), "v"((uint64_t)(uintptr_t)gptr)
                 : "memory");
}

__device__ __forceinline__ void wait_async0() {
    asm volatile("s_wait_asynccnt 0x0" ::: "memory");
}

__device__ __forceinline__ uint32_t lds_off32(const void* p) {
    return (uint32_t)(uintptr_t)p;
}

// ---------------------------------------------------------------------------
// fp32 -> bf16 flat convert (weights)
// ---------------------------------------------------------------------------
__global__ __launch_bounds__(256) void convert_bf16_kernel(
    const float* __restrict__ in, __bf16* __restrict__ out, int n) {
    int i = blockIdx.x * 256 + threadIdx.x;
    if (i < n) out[i] = (__bf16)in[i];
}

// ---------------------------------------------------------------------------
// fp32 [EMB, NPOS] -> bf16 [NPOS, EMB] tiled transpose (LDS 32x33)
// ---------------------------------------------------------------------------
__global__ __launch_bounds__(256) void transpose_to_bf16_kernel(
    const float* __restrict__ in, __bf16* __restrict__ out) {
    __shared__ float tile[32][33];
    int tx = threadIdx.x, ty = threadIdx.y;          // 32 x 8
    int d0 = blockIdx.x * 32;                        // over EMB
    int p0 = blockIdx.y * 32;                        // over NPOS
    for (int k = 0; k < 4; k++) {
        int d = d0 + ty + 8 * k;
        tile[ty + 8 * k][tx] = in[(size_t)d * NPOS + p0 + tx];
    }
    __syncthreads();
    for (int k = 0; k < 4; k++) {
        int p = p0 + ty + 8 * k;
        out[(size_t)p * EMB + d0 + tx] = (__bf16)tile[tx][ty + 8 * k];
    }
}

// ---------------------------------------------------------------------------
// QKV GEMM: C[o,p] = sum_d W[o,d] * act[d,p] + bias[o]; o<1152, p<32768, d<384
// Block tile 128(M) x 64(N). The 64x384 activation B-panel (48 KB, contiguous
// in actT) is staged in LDS ONCE via async load-to-LDS; all 8 waves then read
// B fragments from LDS (was: 8x redundant global loads). A (weights) streams
// from global per-wave. 48 v_wmma per wave.
//   row phase: q,k: [H,S,L,C]; v: [H,S,C,L]
//   col phase: q,k: [H,L,S,C]; v: [H,L,C,S]
// ---------------------------------------------------------------------------
__global__ __launch_bounds__(256) void qkv_gemm_kernel(
    const __bf16* __restrict__ wbf,   // [1152,384]
    const __bf16* __restrict__ actT,  // [32768,384]
    const float*  __restrict__ bias,  // [1152]
    __bf16* __restrict__ qb, __bf16* __restrict__ kb, __bf16* __restrict__ vb,
    int col_phase) {
    __shared__ __bf16 Bl[64 * EMB];   // 49,152 B
    int tid = threadIdx.x;
    int w = tid >> 5, lane = tid & 31;
    int lo = lane & 15;
    int klo = (lane < 16) ? 0 : 8;
    int o0 = blockIdx.x * 128 + w * 16;
    int n0 = blockIdx.y * 64;

    // ---- async stage B panel: 49152 B = 3072 x 16B chunks, 12 per thread ----
    {
        const char* gsrc = (const char*)(actT + (size_t)n0 * EMB);
        uint32_t lbase = lds_off32(&Bl[0]);
        for (int it = 0; it < 12; it++) {
            int chunk = it * 256 + tid;
            async_copy_b128(lbase + chunk * 16, gsrc + (size_t)chunk * 16);
        }
        wait_async0();
    }
    __syncthreads();

    v8f acc[4];
    for (int t = 0; t < 4; t++) acc[t] = vzero8();

    const __bf16* arow = wbf + (size_t)(o0 + lo) * EMB;
    for (int kc = 0; kc < 12; kc++) {
        v16bf a = load_frag(arow + kc * 32, klo);
        for (int t = 0; t < 4; t++) {
            v16bf b = load_frag(&Bl[(size_t)(t * 16 + lo) * EMB + kc * 32], klo);
            acc[t] = wmma_bf16(a, b, acc[t]);
        }
    }

    int mb = (lane >> 4) << 3;  // C-frag: lanes>=16 hold rows M+8
    for (int t = 0; t < 4; t++) {
        int p = n0 + t * 16 + lo;
        int s = p >> 8, l = p & 255;   // p = s*LDIM + l
        for (int r = 0; r < 8; r++) {
            int o = o0 + r + mb;
            float val = acc[t][r] + bias[o];
            int part = o / EMB;        // 0=q 1=k 2=v
            int rem  = o - part * EMB;
            int h = rem >> 5, c = rem & 31;
            __bf16 bv = (__bf16)val;
            if (!col_phase) {
                size_t qk = ((size_t)(h * SDIM + s) * LDIM + l) * CDIM + c;
                if (part == 0)      qb[qk] = bv;
                else if (part == 1) kb[qk] = bv;
                else vb[((size_t)(h * SDIM + s) * CDIM + c) * LDIM + l] = bv;
            } else {
                size_t qk = ((size_t)(h * LDIM + l) * SDIM + s) * CDIM + c;
                if (part == 0)      qb[qk] = bv;
                else if (part == 1) kb[qk] = bv;
                else vb[((size_t)(h * LDIM + l) * CDIM + c) * SDIM + s] = bv;
            }
        }
    }
}

// ---------------------------------------------------------------------------
// Row attention: one block per (h,s). seq = L = 256, head dim C = 32.
// K-tile (256x32 bf16, 16 KB) and V-tile (32x256 bf16, 16 KB) are contiguous
// per (h,s) and staged in LDS via async load-to-LDS; all waves share them.
// scores[i,j] = sum_c q[c,i] k[c,j]; softmax over j; out[d,i] = P[i,:]·v[d,:].
// Output attn_out[(h*32+d), s, i] fp32.  LDS: 16+16+64 = 96 KB.
// ---------------------------------------------------------------------------
__global__ __launch_bounds__(256) void row_attn_kernel(
    const __bf16* __restrict__ qb,  // [H,S,L,C]
    const __bf16* __restrict__ kb,  // [H,S,L,C]
    const __bf16* __restrict__ vb,  // [H,S,C,L]
    float* __restrict__ attn_out) { // [EMB,S,L]
    __shared__ __bf16 Kl[LDIM * CDIM];   // 16 KB
    __shared__ __bf16 Vl[CDIM * LDIM];   // 16 KB
    __shared__ __bf16 Pl[8][16][LDIM];   // 64 KB
    int tid = threadIdx.x;
    int w = tid >> 5, lane = tid & 31;
    int lo = lane & 15, klo = (lane < 16) ? 0 : 8;
    int mb = (lane >> 4) << 3;
    int h = blockIdx.x / SDIM, s = blockIdx.x % SDIM;
    const __bf16* qT = qb + (size_t)(h * SDIM + s) * LDIM * CDIM;
    const __bf16* kT = kb + (size_t)(h * SDIM + s) * LDIM * CDIM;
    const __bf16* vB = vb + (size_t)(h * SDIM + s) * CDIM * LDIM;

    // ---- async stage K and V tiles: 2 x 16384 B = 2 x 1024 chunks ----
    {
        uint32_t kbase = lds_off32(&Kl[0]);
        uint32_t vbase = lds_off32(&Vl[0]);
        const char* kg = (const char*)kT;
        const char* vg = (const char*)vB;
        for (int it = 0; it < 4; it++) {
            int chunk = it * 256 + tid;
            async_copy_b128(kbase + chunk * 16, kg + (size_t)chunk * 16);
            async_copy_b128(vbase + chunk * 16, vg + (size_t)chunk * 16);
        }
        wait_async0();
    }
    __syncthreads();

    for (int half = 0; half < 2; half++) {
        int i0 = w * 32 + half * 16;
        v16bf aq = load_frag(qT + (size_t)(i0 + lo) * CDIM, klo);
        v8f sc[16];
        for (int jt = 0; jt < 16; jt++) {
            v16bf bk = load_frag(&Kl[(size_t)(jt * 16 + lo) * CDIM], klo);
            sc[jt] = wmma_bf16(aq, bk, vzero8());
        }
        // softmax over j (256) per row; row of VGPR r is r (lanes<16) / r+8
        for (int r = 0; r < 8; r++) {
            float m = -3.0e38f;
            for (int jt = 0; jt < 16; jt++) m = fmaxf(m, sc[jt][r]);
            for (int off = 1; off < 16; off <<= 1)
                m = fmaxf(m, __shfl_xor(m, off, 32));
            float e[16];
            float sum = 0.f;
            for (int jt = 0; jt < 16; jt++) { e[jt] = __expf(sc[jt][r] - m); sum += e[jt]; }
            for (int off = 1; off < 16; off <<= 1)
                sum += __shfl_xor(sum, off, 32);
            float inv = 1.f / sum;
            int row = r + mb;
            for (int jt = 0; jt < 16; jt++)
                Pl[w][row][jt * 16 + lo] = (__bf16)(e[jt] * inv);
        }
        // AV: out[i, d] = sum_j P[i,j] v[d,j], d = 0..31 (two N tiles)
        v8f acc0 = vzero8(), acc1 = vzero8();
        for (int kc = 0; kc < 8; kc++) {
            v16bf ap = load_frag(&Pl[w][lo][kc * 32], klo);
            v16bf b0 = load_frag(&Vl[(size_t)(lo) * LDIM + kc * 32], klo);
            v16bf b1 = load_frag(&Vl[(size_t)(16 + lo) * LDIM + kc * 32], klo);
            acc0 = wmma_bf16(ap, b0, acc0);
            acc1 = wmma_bf16(ap, b1, acc1);
        }
        for (int r = 0; r < 8; r++) {
            int i = i0 + r + mb;
            attn_out[((size_t)(h * CDIM + lo)      * SDIM + s) * LDIM + i] = acc0[r];
            attn_out[((size_t)(h * CDIM + 16 + lo) * SDIM + s) * LDIM + i] = acc1[r];
        }
    }
}

// ---------------------------------------------------------------------------
// Column attention: one block per (h,l). seq = S = 128. One 16-row strip per
// wave. K/V tiles (8 KB each) async-staged in LDS. LDS: 8+8+32 = 48 KB.
// Output attn_out[(h*32+d), i, l].
// ---------------------------------------------------------------------------
__global__ __launch_bounds__(256) void col_attn_kernel(
    const __bf16* __restrict__ qb,  // [H,L,S,C]
    const __bf16* __restrict__ kb,  // [H,L,S,C]
    const __bf16* __restrict__ vb,  // [H,L,C,S]
    float* __restrict__ attn_out) { // [EMB,S,L]
    __shared__ __bf16 Kl[SDIM * CDIM];   // 8 KB
    __shared__ __bf16 Vl[CDIM * SDIM];   // 8 KB
    __shared__ __bf16 Pl[8][16][SDIM];   // 32 KB
    int tid = threadIdx.x;
    int w = tid >> 5, lane = tid & 31;
    int lo = lane & 15, klo = (lane < 16) ? 0 : 8;
    int mb = (lane >> 4) << 3;
    int h = blockIdx.x / LDIM, l = blockIdx.x % LDIM;
    const __bf16* qT = qb + (size_t)(h * LDIM + l) * SDIM * CDIM;
    const __bf16* kT = kb + (size_t)(h * LDIM + l) * SDIM * CDIM;
    const __bf16* vB = vb + (size_t)(h * LDIM + l) * CDIM * SDIM;

    // ---- async stage K and V tiles: 2 x 8192 B = 2 x 512 chunks ----
    {
        uint32_t kbase = lds_off32(&Kl[0]);
        uint32_t vbase = lds_off32(&Vl[0]);
        const char* kg = (const char*)kT;
        const char* vg = (const char*)vB;
        for (int it = 0; it < 2; it++) {
            int chunk = it * 256 + tid;
            async_copy_b128(kbase + chunk * 16, kg + (size_t)chunk * 16);
            async_copy_b128(vbase + chunk * 16, vg + (size_t)chunk * 16);
        }
        wait_async0();
    }
    __syncthreads();

    int i0 = w * 16;
    v16bf aq = load_frag(qT + (size_t)(i0 + lo) * CDIM, klo);
    v8f sc[8];
    for (int jt = 0; jt < 8; jt++) {
        v16bf bk = load_frag(&Kl[(size_t)(jt * 16 + lo) * CDIM], klo);
        sc[jt] = wmma_bf16(aq, bk, vzero8());
    }
    for (int r = 0; r < 8; r++) {
        float m = -3.0e38f;
        for (int jt = 0; jt < 8; jt++) m = fmaxf(m, sc[jt][r]);
        for (int off = 1; off < 16; off <<= 1)
            m = fmaxf(m, __shfl_xor(m, off, 32));
        float e[8];
        float sum = 0.f;
        for (int jt = 0; jt < 8; jt++) { e[jt] = __expf(sc[jt][r] - m); sum += e[jt]; }
        for (int off = 1; off < 16; off <<= 1)
            sum += __shfl_xor(sum, off, 32);
        float inv = 1.f / sum;
        int row = r + mb;
        for (int jt = 0; jt < 8; jt++)
            Pl[w][row][jt * 16 + lo] = (__bf16)(e[jt] * inv);
    }
    v8f acc0 = vzero8(), acc1 = vzero8();
    for (int kc = 0; kc < 4; kc++) {
        v16bf ap = load_frag(&Pl[w][lo][kc * 32], klo);
        v16bf b0 = load_frag(&Vl[(size_t)(lo) * SDIM + kc * 32], klo);
        v16bf b1 = load_frag(&Vl[(size_t)(16 + lo) * SDIM + kc * 32], klo);
        acc0 = wmma_bf16(ap, b0, acc0);
        acc1 = wmma_bf16(ap, b1, acc1);
    }
    for (int r = 0; r < 8; r++) {
        int i = i0 + r + mb;
        attn_out[((size_t)(h * CDIM + lo)      * SDIM + i) * LDIM + l] = acc0[r];
        attn_out[((size_t)(h * CDIM + 16 + lo) * SDIM + i) * LDIM + l] = acc1[r];
    }
}

// ---------------------------------------------------------------------------
// Residual + channel LayerNorm: out[d,p] = LN_d(a[d,p] + b[d,p]) * g[d] + beta[d]
// Block = 256 threads, 32 positions; wave w handles channels [48w, 48w+48),
// lanes map to consecutive positions (coalesced). Two-pass with LDS reduce.
// Safe when out aliases a (each index read-then-written by the same thread).
// ---------------------------------------------------------------------------
__global__ __launch_bounds__(256) void ln_residual_kernel(
    const float* __restrict__ a, const float* __restrict__ b,
    const float* __restrict__ gamma, const float* __restrict__ beta,
    float* __restrict__ out) {
    __shared__ float psum[8][32], psq[8][32], mu_s[32], rs_s[32];
    int w = threadIdx.x >> 5, lane = threadIdx.x & 31;
    int p = blockIdx.x * 32 + lane;
    float sum = 0.f, sq = 0.f;
    for (int t = 0; t < 48; t++) {
        int d = w * 48 + t;
        size_t idx = (size_t)d * NPOS + p;
        float v = a[idx] + b[idx];
        sum += v; sq += v * v;
    }
    psum[w][lane] = sum; psq[w][lane] = sq;
    __syncthreads();
    if (w == 0) {
        float S = 0.f, Q = 0.f;
        for (int j = 0; j < 8; j++) { S += psum[j][lane]; Q += psq[j][lane]; }
        float mu = S / (float)EMB;
        float var = Q / (float)EMB - mu * mu;
        mu_s[lane] = mu;
        rs_s[lane] = rsqrtf(var + LN_EPS);
    }
    __syncthreads();
    float mu = mu_s[lane], rs = rs_s[lane];
    for (int t = 0; t < 48; t++) {
        int d = w * 48 + t;
        size_t idx = (size_t)d * NPOS + p;
        float v = a[idx] + b[idx];
        out[idx] = (v - mu) * rs * gamma[d] + beta[d];
    }
}

// ---------------------------------------------------------------------------
// Host-side orchestration (graph-capture safe: launches only, all on stream).
// Workspace layout (bytes):
//   [0)            w_bf16   1152*384*2      =    884,736
//   [884736)       actT     32768*384*2     = 25,165,824
//   [26050560)     q        25,165,824
//   [51216384)     k        25,165,824
//   [76382208)     v        25,165,824
//   [101548032)    attn_out 384*32768*4     = 50,331,648   (end ~151.9 MB)
// d_out doubles as the post-LN1 activation for phase 2.
// ---------------------------------------------------------------------------
extern "C" void kernel_launch(void* const* d_in, const int* in_sizes, int n_in,
                              void* d_out, int out_size, void* d_ws, size_t ws_size,
                              hipStream_t stream) {
    (void)in_sizes; (void)n_in; (void)out_size; (void)ws_size;
    const float* x     = (const float*)d_in[0];
    const float* row_w = (const float*)d_in[1];
    const float* row_b = (const float*)d_in[2];
    const float* col_w = (const float*)d_in[3];
    const float* col_b = (const float*)d_in[4];
    const float* ln1w  = (const float*)d_in[5];
    const float* ln1b  = (const float*)d_in[6];
    const float* ln2w  = (const float*)d_in[7];
    const float* ln2b  = (const float*)d_in[8];
    float* out = (float*)d_out;

    char* ws = (char*)d_ws;
    __bf16* wbf  = (__bf16*)(ws);
    __bf16* actT = (__bf16*)(ws + 884736);
    __bf16* qb   = (__bf16*)(ws + 26050560);
    __bf16* kb   = (__bf16*)(ws + 51216384);
    __bf16* vb   = (__bf16*)(ws + 76382208);
    float*  attn = (float*)(ws + 101548032);

    dim3 blk256(256);
    dim3 tgrid(EMB / 32, NPOS / 32);
    dim3 tblk(32, 8);
    dim3 ggrid(K3 / 128, NPOS / 64);

    // ---- phase 1: row attention ----
    convert_bf16_kernel<<<dim3((K3 * EMB + 255) / 256), blk256, 0, stream>>>(row_w, wbf, K3 * EMB);
    transpose_to_bf16_kernel<<<tgrid, tblk, 0, stream>>>(x, actT);
    qkv_gemm_kernel<<<ggrid, blk256, 0, stream>>>(wbf, actT, row_b, qb, kb, vb, 0);
    row_attn_kernel<<<dim3(HEADS * SDIM), blk256, 0, stream>>>(qb, kb, vb, attn);
    ln_residual_kernel<<<dim3(NPOS / 32), blk256, 0, stream>>>(x, attn, ln1w, ln1b, out);

    // ---- phase 2: column attention ----
    convert_bf16_kernel<<<dim3((K3 * EMB + 255) / 256), blk256, 0, stream>>>(col_w, wbf, K3 * EMB);
    transpose_to_bf16_kernel<<<tgrid, tblk, 0, stream>>>(out, actT);
    qkv_gemm_kernel<<<ggrid, blk256, 0, stream>>>(wbf, actT, col_b, qb, kb, vb, 1);
    col_attn_kernel<<<dim3(HEADS * LDIM), blk256, 0, stream>>>(qb, kb, vb, attn);
    ln_residual_kernel<<<dim3(NPOS / 32), blk256, 0, stream>>>(out, attn, ln2w, ln2b, out);
}
